// SeaReader_36687610643015
// MI455X (gfx1250) — matrix operand order repaired and checked by reading, not compiled, not run on hardware
//
// SeaReader forward for MI455X (gfx1250, wave32, WMMA).
// All GEMMs: bf16 A/B, f32 accumulate via v_wmma_f32_16x16x32_bf16.
// Self-attention is fused flash-style (no 256MB score materialization).
// Data movement: CDNA5 async Global->LDS (global_load_async_to_lds_b128,
// ASYNCcnt) with double-buffered tiles in the GEMMs and attention kernel.
// Workspace: linear bump allocation, ~410MB assumed available in d_ws.
#include <hip/hip_runtime.h>
#include <hip/hip_bf16.h>
#include <math.h>

typedef __bf16 bf16_t;
typedef __attribute__((ext_vector_type(16))) __bf16 v16bf;
typedef __attribute__((ext_vector_type(8)))  __bf16 v8bf;
typedef __attribute__((ext_vector_type(8)))  float  v8f;

#define DEVINL __device__ __forceinline__

DEVINL float sigmf(float x){ return 1.f/(1.f+__expf(-x)); }

DEVINL v8f wmma_bf16(v16bf a, v16bf b, v8f c){
  // D = A(16x32) * B(32x16) + C, f32 accum
  return __builtin_amdgcn_wmma_f32_16x16x32_bf16(false, a, false, b, (short)0, c,
                                                 false, false);
}

// ---- CDNA5 async Global->LDS copy (ISA 15.18.3 op 98, ASYNCcnt) ----
// LDS dest address = low 32 bits of the flat pointer (ISA 10.2: LDS aperture
// addresses carry the LDS byte offset in addr[31:0]).
DEVINL void async_ld_b128(void* lds, const void* gptr){
  unsigned off = (unsigned)(size_t)lds;
  asm volatile("global_load_async_to_lds_b128 %0, %1, off"
               :: "v"(off), "v"(gptr) : "memory");
}
#define WAIT_ASYNC(imm) asm volatile("s_wait_asynccnt " imm ::: "memory")

// ---- WMMA fragment loaders from row-major LDS tiles (contiguous k = 16*hi + e, same for A and B) ----
// A 16x32 bf16: lane m=lane&15, hi=lane>>4; elem e -> K=hi*16+e
DEVINL v16bf frag_a(const bf16_t* tile, int ldk, int row0, int k0, int lane){
  int m  = row0 + (lane & 15);
  int hi = (lane >> 4) & 1;
  const bf16_t* p = tile + (size_t)m*ldk + k0 + hi*16;
  v8bf lo = *(const v8bf*)p;
  v8bf hh = *(const v8bf*)(p + 8);
  return __builtin_shufflevector(lo, hh, 0,1,2,3,4,5,6,7,8,9,10,11,12,13,14,15);
}
// B 32x16 bf16 from an [N][K] row-major tile (computes A*B^T): Bm[kk][n]=tile[n][k0+kk]
DEVINL v16bf frag_b_nt(const bf16_t* tile, int ldk, int n0, int k0, int lane){
  int n  = n0 + (lane & 15);
  int hi = (lane >> 4) & 1;
  return *(const v16bf*)(tile + (size_t)n*ldk + k0 + hi*16);
}
// B 32x16 bf16 from a [K][N] row-major tile: Bm[kk][n]=tile[k0+kk][n0+n] (gather)
DEVINL v16bf frag_b_nn(const bf16_t* tile, int ldn, int k0, int n0, int lane){
  int n  = n0 + (lane & 15);
  int hi = (lane >> 4) & 1;
  v16bf b{};
  #pragma unroll
  for (int e=0;e<16;++e) b[e] = tile[(size_t)(k0 + e + hi*16)*ldn + n];
  return b;
}

// batch-index remapping: n = ((b*4+opt)*8+topn); statement seq = n>>3; doc seq = b*8+topn
DEVINL long map_idx(int mode, int z){
  if (mode == 1) return (long)(z >> 3);                    // unique statement
  if (mode == 2) return (long)(((z >> 5) << 3) | (z & 7)); // unique document
  return (long)z;
}

// ---------------- generic converts / builders ----------------
__global__ void k_f2bf(const float* __restrict__ in, bf16_t* __restrict__ out, long n){
  long i = (long)blockIdx.x*blockDim.x + threadIdx.x;
  if (i < n) out[i] = (bf16_t)in[i];
}

// stat[b,o,t,v]: t<32 -> statement[b,t,v] else option[b,o,t-32,v]; out bf16 [16,64,256]
__global__ void k_build_stat(const float* __restrict__ st, const float* __restrict__ op,
                             bf16_t* __restrict__ out){
  long i = (long)blockIdx.x*blockDim.x + threadIdx.x;
  if (i >= 16L*64*256) return;
  int v = (int)(i & 255);
  long r = i >> 8;
  int tp = (int)(r & 63);
  int bo = (int)(r >> 6);
  int b = bo >> 2, o = bo & 3;
  float val = (tp < 32) ? st[((size_t)b*32 + tp)*256 + v]
                        : op[(((size_t)(b*4 + o))*32 + (tp-32))*256 + v];
  out[i] = (bf16_t)val;
}

// ---------------- batched WMMA GEMMs, 64x64 tile, 8 waves, async double-buffer ----------------
// C[z] = A[map(z)] * B[map(z)]^T (+bias); A:[M,K] B:[N,K] bf16 row-major, C f32
__global__ void __launch_bounds__(256)
k_gemm_nt(const bf16_t* __restrict__ A, long sA, int amap,
          const bf16_t* __restrict__ B, long sB, int bmap,
          float* __restrict__ C, long sC, const float* __restrict__ bias,
          int M, int N, int K){
  __shared__ alignas(32) bf16_t As[2][64][32];
  __shared__ alignas(32) bf16_t Bs[2][64][32];
  (void)M;
  int m0 = blockIdx.x*64, n0 = blockIdx.y*64, z = blockIdx.z;
  const bf16_t* Ab = A + map_idx(amap, z)*sA;
  const bf16_t* Bb = B + map_idx(bmap, z)*sB;
  float* Cb = C + (size_t)z*sC;
  int t = threadIdx.x, w = t>>5, lane = t&31;
  int wm = (w>>1)*16, wn = (w&1)*32;
  int row = t>>2, c8 = (t&3)*8;                 // 16B chunk per thread per tile
  auto issue = [&](int buf, int k0){
    async_ld_b128(&As[buf][row][c8], Ab + (size_t)(m0+row)*K + k0 + c8);
    async_ld_b128(&Bs[buf][row][c8], Bb + (size_t)(n0+row)*K + k0 + c8);
  };
  issue(0, 0);
  v8f acc0 = {}, acc1 = {};
  int cur = 0;
  for (int k0=0; k0<K; k0+=32, cur^=1){
    if (k0+32 < K){ issue(cur^1, k0+32); WAIT_ASYNC("0x2"); }
    else          { WAIT_ASYNC("0x0"); }
    __syncthreads();
    v16bf a  = frag_a(&As[cur][0][0], 32, wm, 0, lane);
    v16bf b0 = frag_b_nt(&Bs[cur][0][0], 32, wn,    0, lane);
    v16bf b1 = frag_b_nt(&Bs[cur][0][0], 32, wn+16, 0, lane);
    acc0 = wmma_bf16(a, b0, acc0);
    acc1 = wmma_bf16(a, b1, acc1);
    __syncthreads();
  }
  int nc = lane & 15, hi = (lane>>4)&1;
  #pragma unroll
  for (int j=0;j<8;++j){
    int r  = m0 + wm + j + 8*hi;
    int c0 = n0 + wn + nc, c1 = n0 + wn + 16 + nc;
    Cb[(size_t)r*N + c0] = acc0[j] + (bias ? bias[c0] : 0.f);
    Cb[(size_t)r*N + c1] = acc1[j] + (bias ? bias[c1] : 0.f);
  }
}

// C[z] = A[map(z)] * B[map(z)]; A:[M,K] bf16, B:[K,N] bf16 row-major, C f32
__global__ void __launch_bounds__(256)
k_gemm_nn(const bf16_t* __restrict__ A, long sA, int amap,
          const bf16_t* __restrict__ B, long sB, int bmap,
          float* __restrict__ C, long sC, int M, int N, int K){
  __shared__ alignas(32) bf16_t As[2][64][32];
  __shared__ alignas(32) bf16_t Bs[2][32][64];
  (void)M;
  int m0 = blockIdx.x*64, n0 = blockIdx.y*64, z = blockIdx.z;
  const bf16_t* Ab = A + map_idx(amap, z)*sA;
  const bf16_t* Bb = B + map_idx(bmap, z)*sB;
  float* Cb = C + (size_t)z*sC;
  int t = threadIdx.x, w = t>>5, lane = t&31;
  int wm = (w>>1)*16, wn = (w&1)*32;
  int row = t>>2, c8 = (t&3)*8;
  int br = t>>3, bc8 = (t&7)*8;
  auto issue = [&](int buf, int k0){
    async_ld_b128(&As[buf][row][c8], Ab + (size_t)(m0+row)*K + k0 + c8);
    async_ld_b128(&Bs[buf][br][bc8], Bb + (size_t)(k0+br)*N + n0 + bc8);
  };
  issue(0, 0);
  v8f acc0 = {}, acc1 = {};
  int cur = 0;
  for (int k0=0; k0<K; k0+=32, cur^=1){
    if (k0+32 < K){ issue(cur^1, k0+32); WAIT_ASYNC("0x2"); }
    else          { WAIT_ASYNC("0x0"); }
    __syncthreads();
    v16bf a  = frag_a(&As[cur][0][0], 32, wm, 0, lane);
    v16bf b0 = frag_b_nn(&Bs[cur][0][0], 64, 0, wn,    lane);
    v16bf b1 = frag_b_nn(&Bs[cur][0][0], 64, 0, wn+16, lane);
    acc0 = wmma_bf16(a, b0, acc0);
    acc1 = wmma_bf16(a, b1, acc1);
    __syncthreads();
  }
  int nc = lane & 15, hi = (lane>>4)&1;
  #pragma unroll
  for (int j=0;j<8;++j){
    int r = m0 + wm + j + 8*hi;
    Cb[(size_t)r*N + n0 + wn + nc]      = acc0[j];
    Cb[(size_t)r*N + n0 + wn + 16 + nc] = acc1[j];
  }
}

// ---------------- GRU recurrence: one block per sequence per direction ----------------
// xp:[nseq,T,384] precomputed input proj (+bih); whh:[384,128] f32; out hcat:[nseq,T,256]
__global__ void __launch_bounds__(384)
k_gru(const float* __restrict__ xp, const float* __restrict__ whh,
      const float* __restrict__ bhh, float* __restrict__ hout,
      int T, int rev, int off){
  __shared__ float h[128];
  __shared__ float gh[384];
  int seq = blockIdx.x;
  int t = threadIdx.x;
  float wrow[128];                   // register-resident whh row
  #pragma unroll
  for (int k=0;k<128;++k) wrow[k] = whh[(size_t)t*128 + k];
  float bh = bhh[t];
  if (t < 128) h[t] = 0.f;
  __syncthreads();
  for (int s=0; s<T; ++s){
    int tt = rev ? (T-1-s) : s;
    float acc = bh;
    #pragma unroll
    for (int k=0;k<128;++k) acc += h[k]*wrow[k];
    gh[t] = acc;
    __syncthreads();
    if (t < 128){
      const float* x = xp + ((size_t)seq*T + tt)*384;
      float r  = sigmf(x[t]       + gh[t]);
      float zz = sigmf(x[128+t]   + gh[128+t]);
      float nn = tanhf(x[256+t] + r*gh[256+t]);
      float hn = (1.f - zz)*nn + zz*h[t];
      h[t] = hn;
      hout[((size_t)seq*T + tt)*256 + off + t] = hn;
    }
    __syncthreads();
  }
}

// ---------------- softmaxes ----------------
__global__ void __launch_bounds__(256)
k_rowsoftmax(const float* __restrict__ X, bf16_t* __restrict__ P){
  __shared__ float red[256];
  size_t row = blockIdx.x; int t = threadIdx.x;
  float v = X[row*256 + t];
  red[t] = v; __syncthreads();
  for (int s2=128;s2>0;s2>>=1){ if (t<s2) red[t]=fmaxf(red[t],red[t+s2]); __syncthreads(); }
  float mx = red[0]; __syncthreads();
  float p = __expf(v - mx);
  red[t] = p; __syncthreads();
  for (int s2=128;s2>0;s2>>=1){ if (t<s2) red[t]+=red[t+s2]; __syncthreads(); }
  P[row*256 + t] = (bf16_t)(p / red[0]);
}

// column softmax over s=64 of match[n,64,256], transposed out PT[n,256,64] bf16
__global__ void k_colsoftmaxT(const float* __restrict__ M, bf16_t* __restrict__ PT){
  int idx = blockIdx.x*blockDim.x + threadIdx.x;
  if (idx >= 128*256) return;
  int n = idx >> 8, d = idx & 255;
  const float* col = M + (size_t)n*64*256 + d;
  float buf[64];
  float mx = -3.0e38f;
  #pragma unroll
  for (int s=0;s<64;++s){ buf[s] = col[(size_t)s*256]; mx = fmaxf(mx, buf[s]); }
  float sum = 0.f;
  #pragma unroll
  for (int s=0;s<64;++s){ buf[s] = __expf(buf[s]-mx); sum += buf[s]; }
  float inv = 1.f/sum;
  bf16_t* o = PT + ((size_t)n*256 + d)*64;
  #pragma unroll
  for (int s=0;s<64;++s) o[s] = (bf16_t)(buf[s]*inv);
}

// doc_read concat -> bf16 [128*256, 512]: [0:256]=docstmp_u[dix], [256:512]=doc_att[n]
__global__ void k_concat(const float* __restrict__ docstmp, const float* __restrict__ docatt,
                         bf16_t* __restrict__ dr){
  long i = (long)blockIdx.x*blockDim.x + threadIdx.x;
  if (i >= 128L*256*512) return;
  int e = (int)(i & 511);
  long td = i >> 9;
  int n = (int)(td >> 8), d = (int)(td & 255);
  float v;
  if (e < 256){
    int dix = ((n>>5)<<3) | (n&7);
    v = docstmp[((size_t)dix*256 + d)*256 + e];
  } else {
    v = docatt[((size_t)n*256 + d)*256 + (e-256)];
  }
  dr[i] = (bf16_t)v;
}

// ---------------- fused flash self-attention over [16, 2048, 512] ----------------
// Q tile 64x512 + double-buffered K/V tiles (async Global->LDS), online softmax,
// O accumulated in 128 VGPRs/wave. ~217KB LDS (WGP has 320KB).
__global__ void __launch_bounds__(256)
k_selfattn(const bf16_t* __restrict__ drb, float* __restrict__ out){
  __shared__ alignas(32) bf16_t Qs[64][512];
  __shared__ alignas(32) bf16_t Ks[2][64][512];
  __shared__ float Ss[64][64];
  __shared__ alignas(32) bf16_t Ps[64][64];
  __shared__ float row_m[64], row_l[64], row_sc[64];
  int qt = blockIdx.x, bz = blockIdx.y;
  const bf16_t* base = drb + (size_t)bz*2048*512;
  int t = threadIdx.x, w = t>>5, lane = t&31;
  int wm = (w>>1)*16, wnh = (w&1)*256;
  // async preload: Q tile + K tile 0  (16 b128 chunks per thread each)
  for (int i=t; i<4096; i+=256){
    int r = i>>6, c8 = (i&63)*8;
    async_ld_b128(&Qs[r][c8], base + (size_t)(qt*64+r)*512 + c8);
  }
  for (int i=t; i<4096; i+=256){
    int r = i>>6, c8 = (i&63)*8;
    async_ld_b128(&Ks[0][r][c8], base + (size_t)r*512 + c8);
  }
  if (t < 64){ row_m[t] = -3.0e38f; row_l[t] = 0.f; }
  v8f acc[16] = {};                                    // O: 16x256 f32 per wave
  int cur = 0;
  for (int kt=0; kt<32; ++kt, cur^=1){
    if (kt+1 < 32){
      for (int i=t; i<4096; i+=256){                   // prefetch next K/V tile
        int r = i>>6, c8 = (i&63)*8;
        async_ld_b128(&Ks[cur^1][r][c8], base + (size_t)((kt+1)*64+r)*512 + c8);
      }
      WAIT_ASYNC("0x10");                              // current tile (+Q) landed
    } else {
      WAIT_ASYNC("0x0");
    }
    __syncthreads();
    // S = Q * K^T  (64x64 in 16 wmma tiles; 2 per wave)
    #pragma unroll
    for (int rep=0; rep<2; ++rep){
      int id = w*2 + rep, si = id>>2, sj = id&3;
      v8f s = {};
      for (int k0=0; k0<512; k0+=32){
        v16bf a = frag_a(&Qs[0][0], 512, si*16, k0, lane);
        v16bf b = frag_b_nt(&Ks[cur][0][0], 512, sj*16, k0, lane);
        s = wmma_bf16(a, b, s);
      }
      int nc = lane & 15, hi = (lane>>4)&1;
      #pragma unroll
      for (int j=0;j<8;++j) Ss[si*16 + j + 8*hi][sj*16 + nc] = s[j];
    }
    __syncthreads();
    // online softmax update (waves 0-1; uniform per wave -> no WMMA divergence)
    if (t < 64){
      float m = row_m[t], mx = m;
      for (int j=0;j<64;++j) mx = fmaxf(mx, Ss[t][j]);
      float sc = __expf(m - mx);
      float l  = row_l[t]*sc;
      for (int j=0;j<64;++j){
        float p = __expf(Ss[t][j] - mx);
        l += p;
        Ps[t][j] = (bf16_t)p;
      }
      row_m[t]=mx; row_l[t]=l; row_sc[t]=sc;
    }
    __syncthreads();
    // rescale O and accumulate P * V
    int hi = (lane>>4)&1;
    float scj[8];
    #pragma unroll
    for (int j=0;j<8;++j) scj[j] = row_sc[wm + j + 8*hi];
    #pragma unroll
    for (int nt2=0; nt2<16; ++nt2){
      #pragma unroll
      for (int j=0;j<8;++j) acc[nt2][j] *= scj[j];
    }
    #pragma unroll
    for (int k0=0; k0<64; k0+=32){
      v16bf a = frag_a(&Ps[0][0], 64, wm, k0, lane);
      for (int nt2=0; nt2<16; ++nt2){
        v16bf b = frag_b_nn(&Ks[cur][0][0], 512, k0, wnh + nt2*16, lane);
        acc[nt2] = wmma_bf16(a, b, acc[nt2]);
      }
    }
    __syncthreads();   // protect Ks[cur] before next iteration's prefetch of it
  }
  int nc = lane & 15, hi = (lane>>4)&1;
  float invl[8];
  #pragma unroll
  for (int j=0;j<8;++j) invl[j] = 1.f/row_l[wm + j + 8*hi];
  for (int nt2=0; nt2<16; ++nt2){
    #pragma unroll
    for (int j=0;j<8;++j){
      int r = qt*64 + wm + j + 8*hi;
      int c = wnh + nt2*16 + nc;
      out[((size_t)bz*2048 + r)*512 + c] = acc[nt2][j]*invl[j];
    }
  }
}

// ---------------- sigmoid scalar gate: x *= sigmoid(x.w + b); also emit bf16 ----------------
__global__ void k_gate(float* __restrict__ X, bf16_t* __restrict__ Xb,
                       const float* __restrict__ w, const float* __restrict__ b, int E){
  extern __shared__ float red[];
  size_t tok = blockIdx.x; int t = threadIdx.x;
  float x = X[tok*E + t];
  red[t] = x*w[t]; __syncthreads();
  for (int s2=E>>1; s2>0; s2>>=1){ if (t<s2) red[t]+=red[t+s2]; __syncthreads(); }
  float g = sigmf(red[0] + b[0]);
  float y = g*x;
  X[tok*E + t]  = y;
  Xb[tok*E + t] = (bf16_t)y;
}

__global__ void k_maxpool(const float* __restrict__ H, float* __restrict__ O, int T){
  int n = blockIdx.x, c = threadIdx.x;   // C == 256
  float m = -3.0e38f;
  for (int s=0;s<T;++s) m = fmaxf(m, H[((size_t)n*T + s)*256 + c]);
  O[(size_t)n*256 + c] = m;
}

// feat=[stmt_max, docs_max]; feat*= (feat.integ_w + b); max/mean over topn; out = final.out_w + b
__global__ void __launch_bounds__(512)
k_final(const float* __restrict__ smax, const float* __restrict__ dmax,
        const float* __restrict__ iw, const float* __restrict__ ib,
        const float* __restrict__ ow, const float* __restrict__ ob,
        float* __restrict__ out){
  __shared__ float red[512];
  int z = blockIdx.x; int e = threadIdx.x;
  float fmx = -3.0e38f, fsum = 0.f;
  float iwe = iw[e];
  for (int i=0;i<8;++i){
    int n = z*8 + i;
    float f = (e < 256) ? smax[(size_t)n*256 + e] : dmax[(size_t)n*256 + (e-256)];
    red[e] = f*iwe; __syncthreads();
    for (int s2=256;s2>0;s2>>=1){ if (e<s2) red[e]+=red[e+s2]; __syncthreads(); }
    float g = red[0] + ib[0];
    __syncthreads();
    float fg = f*g;
    fmx = fmaxf(fmx, fg); fsum += fg;
  }
  float contrib = fmx*ow[e] + (fsum*0.125f)*ow[512+e];
  red[e] = contrib; __syncthreads();
  for (int s2=256;s2>0;s2>>=1){ if (e<s2) red[e]+=red[e+s2]; __syncthreads(); }
  if (e == 0) out[z] = red[0] + ob[0];
}

// ---------------- host side ----------------
enum {
  IN_STATEMENT=0, IN_OPTIONS=1, IN_DOCS=2,
  CTX_WIH_F=3, CTX_WHH_F, CTX_BIH_F, CTX_BHH_F, CTX_WIH_B, CTX_WHH_B, CTX_BIH_B, CTX_BHH_B,
  ST_WIH_F=11, ST_WHH_F, ST_BIH_F, ST_BHH_F, ST_WIH_B, ST_WHH_B, ST_BIH_B, ST_BHH_B,
  DOC_WIH_F=19, DOC_WHH_F, DOC_BIH_F, DOC_BHH_F, DOC_WIH_B, DOC_WHH_B, DOC_BIH_B, DOC_BHH_B,
  GATE_S_W=27, GATE_S_B, GATE_D_W, GATE_D_B, INTEG_W, INTEG_B, OUT_W, OUT_B
};

extern "C" void kernel_launch(void* const* d_in, const int* in_sizes, int n_in,
                              void* d_out, int out_size, void* d_ws, size_t ws_size,
                              hipStream_t stream){
  (void)in_sizes; (void)n_in; (void)out_size; (void)ws_size;
  auto fin = [&](int i){ return (const float*)d_in[i]; };

  char* ws = (char*)d_ws;
  size_t off = 0;
  auto alloc = [&](size_t bytes)->char*{
    char* p = ws + off;
    off = (off + bytes + 255) & ~(size_t)255;
    return p;
  };

  // bf16 weight copies (B operands, NT)
  bf16_t* wctx_f = (bf16_t*)alloc(384*256*2);
  bf16_t* wctx_b = (bf16_t*)alloc(384*256*2);
  bf16_t* wst_f  = (bf16_t*)alloc(384*256*2);
  bf16_t* wst_b  = (bf16_t*)alloc(384*256*2);
  bf16_t* wdoc_f = (bf16_t*)alloc(384*512*2);
  bf16_t* wdoc_b = (bf16_t*)alloc(384*512*2);
  bf16_t* stat_bf    = (bf16_t*)alloc(16L*64*256*2);
  bf16_t* docs_bf    = (bf16_t*)alloc(32L*256*256*2);
  float*  xp_stat_f  = (float*)alloc(1024L*384*4);
  float*  xp_stat_b  = (float*)alloc(1024L*384*4);
  float*  xp_docs_f  = (float*)alloc(8192L*384*4);
  float*  xp_docs_b  = (float*)alloc(8192L*384*4);
  float*  statmp     = (float*)alloc(16L*64*256*4);     // unique statements [16,64,256]
  float*  docstmp    = (float*)alloc(32L*256*256*4);    // unique docs [32,256,256]
  bf16_t* statmp_bf  = (bf16_t*)alloc(16L*64*256*2);
  bf16_t* docstmp_bf = (bf16_t*)alloc(32L*256*256*2);
  float*  match      = (float*)alloc(128L*64*256*4);
  bf16_t* p_bf       = (bf16_t*)alloc(128L*64*256*2);
  bf16_t* pt_bf      = (bf16_t*)alloc(128L*256*64*2);
  float*  rsum       = (float*)alloc(128L*64*256*4);    // read_summary -> gated stmt
  bf16_t* stmt_bf    = (bf16_t*)alloc(128L*64*256*2);
  float*  docatt     = (float*)alloc(128L*256*256*4);
  bf16_t* dr_bf      = (bf16_t*)alloc(128L*256*512*2);  // self-attn Q=K=V
  float*  dr2        = (float*)alloc(128L*256*512*4);   // self-attn output
  bf16_t* docsg_bf   = (bf16_t*)alloc(128L*256*512*2);
  float*  xp_st_f    = (float*)alloc(8192L*384*4);
  float*  xp_st_b    = (float*)alloc(8192L*384*4);
  float*  xp_doc_f   = (float*)alloc(32768L*384*4);
  float*  xp_doc_b   = (float*)alloc(32768L*384*4);
  float*  stmt_h     = (float*)alloc(128L*64*256*4);
  float*  docsg_h    = (float*)alloc(128L*256*256*4);
  float*  stmt_max   = (float*)alloc(128L*256*4);
  float*  docs_max   = (float*)alloc(128L*256*4);

  auto cvt = [&](const float* src, bf16_t* dst, long n){
    unsigned g = (unsigned)((n + 255)/256);
    k_f2bf<<<g,256,0,stream>>>(src, dst, n);
  };

  // 0) weights -> bf16
  cvt(fin(CTX_WIH_F), wctx_f, 384L*256);
  cvt(fin(CTX_WIH_B), wctx_b, 384L*256);
  cvt(fin(ST_WIH_F),  wst_f,  384L*256);
  cvt(fin(ST_WIH_B),  wst_b,  384L*256);
  cvt(fin(DOC_WIH_F), wdoc_f, 384L*512);
  cvt(fin(DOC_WIH_B), wdoc_b, 384L*512);

  // 1) build inputs (dedup: 16 statement seqs, 32 doc seqs)
  k_build_stat<<<1024,256,0,stream>>>(fin(IN_STATEMENT), fin(IN_OPTIONS), stat_bf);
  cvt(fin(IN_DOCS), docs_bf, 32L*256*256);

  // 2) ctx BiGRU input projections (WMMA GEMM, +bih)
  k_gemm_nt<<<dim3(16,6,1),256,0,stream>>>(stat_bf,0,0, wctx_f,0,0, xp_stat_f,0,
                                           fin(CTX_BIH_F), 1024,384,256);
  k_gemm_nt<<<dim3(16,6,1),256,0,stream>>>(stat_bf,0,0, wctx_b,0,0, xp_stat_b,0,
                                           fin(CTX_BIH_B), 1024,384,256);
  k_gemm_nt<<<dim3(128,6,1),256,0,stream>>>(docs_bf,0,0, wctx_f,0,0, xp_docs_f,0,
                                            fin(CTX_BIH_F), 8192,384,256);
  k_gemm_nt<<<dim3(128,6,1),256,0,stream>>>(docs_bf,0,0, wctx_b,0,0, xp_docs_b,0,
                                            fin(CTX_BIH_B), 8192,384,256);

  // 3) ctx GRU recurrences
  k_gru<<<16,384,0,stream>>>(xp_stat_f, fin(CTX_WHH_F), fin(CTX_BHH_F), statmp, 64, 0, 0);
  k_gru<<<16,384,0,stream>>>(xp_stat_b, fin(CTX_WHH_B), fin(CTX_BHH_B), statmp, 64, 1, 128);
  k_gru<<<32,384,0,stream>>>(xp_docs_f, fin(CTX_WHH_F), fin(CTX_BHH_F), docstmp, 256, 0, 0);
  k_gru<<<32,384,0,stream>>>(xp_docs_b, fin(CTX_WHH_B), fin(CTX_BHH_B), docstmp, 256, 1, 128);
  cvt(statmp, statmp_bf, 16L*64*256);
  cvt(docstmp, docstmp_bf, 32L*256*256);

  // 4) co-attention: match = stat @ doc^T (batched over n=128 with dedup remap)
  k_gemm_nt<<<dim3(1,4,128),256,0,stream>>>(statmp_bf, 64L*256, 1,
                                            docstmp_bf, 256L*256, 2,
                                            match, 64L*256, nullptr, 64,256,256);
  k_rowsoftmax<<<8192,256,0,stream>>>(match, p_bf);
  k_colsoftmaxT<<<128,256,0,stream>>>(match, pt_bf);
  k_gemm_nn<<<dim3(1,4,128),256,0,stream>>>(p_bf, 64L*256, 0,
                                            docstmp_bf, 256L*256, 2,
                                            rsum, 64L*256, 64,256,256);
  k_gemm_nn<<<dim3(4,4,128),256,0,stream>>>(pt_bf, 256L*64, 0,
                                            statmp_bf, 64L*256, 1,
                                            docatt, 256L*256, 256,256,64);

  // 5) doc_read = [docstmp, doc_att] -> bf16, then fused self-attention
  k_concat<<<65536,256,0,stream>>>(docstmp, docatt, dr_bf);
  k_selfattn<<<dim3(32,16),256,0,stream>>>(dr_bf, dr2);

  // 6) sigmoid gates
  k_gate<<<32768,512,512*4,stream>>>(dr2,  docsg_bf, fin(GATE_D_W), fin(GATE_D_B), 512);
  k_gate<<<8192, 256,256*4,stream>>>(rsum, stmt_bf,  fin(GATE_S_W), fin(GATE_S_B), 256);

  // 7) st / doc BiGRU projections + recurrences
  k_gemm_nt<<<dim3(128,6,1),256,0,stream>>>(stmt_bf,0,0, wst_f,0,0, xp_st_f,0,
                                            fin(ST_BIH_F), 8192,384,256);
  k_gemm_nt<<<dim3(128,6,1),256,0,stream>>>(stmt_bf,0,0, wst_b,0,0, xp_st_b,0,
                                            fin(ST_BIH_B), 8192,384,256);
  k_gemm_nt<<<dim3(512,6,1),256,0,stream>>>(docsg_bf,0,0, wdoc_f,0,0, xp_doc_f,0,
                                            fin(DOC_BIH_F), 32768,384,512);
  k_gemm_nt<<<dim3(512,6,1),256,0,stream>>>(docsg_bf,0,0, wdoc_b,0,0, xp_doc_b,0,
                                            fin(DOC_BIH_B), 32768,384,512);
  k_gru<<<128,384,0,stream>>>(xp_st_f,  fin(ST_WHH_F),  fin(ST_BHH_F),  stmt_h,  64, 0, 0);
  k_gru<<<128,384,0,stream>>>(xp_st_b,  fin(ST_WHH_B),  fin(ST_BHH_B),  stmt_h,  64, 1, 128);
  k_gru<<<128,384,0,stream>>>(xp_doc_f, fin(DOC_WHH_F), fin(DOC_BHH_F), docsg_h, 256, 0, 0);
  k_gru<<<128,384,0,stream>>>(xp_doc_b, fin(DOC_WHH_B), fin(DOC_BHH_B), docsg_h, 256, 1, 128);

  // 8) maxpool over time, integrate gate + topn max/mean + readout
  k_maxpool<<<128,256,0,stream>>>(stmt_h,  stmt_max, 64);
  k_maxpool<<<128,256,0,stream>>>(docsg_h, docs_max, 256);
  k_final<<<16,512,0,stream>>>(stmt_max, docs_max, fin(INTEG_W), fin(INTEG_B),
                               fin(OUT_W), fin(OUT_B), (float*)d_out);
}
